// STDIMModel_64003602645552
// MI455X (gfx1250) — matrix operand order repaired and checked by reading, not compiled, not run on hardware
//
#include <hip/hip_runtime.h>
#include <hip/hip_bf16.h>
#include <math.h>

// Problem constants (from reference): N=512 batch, D=256 feat, C=128 chan, L=12*12=144 locations.
#define LN  144
#define NN  512
#define CC  128
#define DD  256

typedef _Float16 v16h __attribute__((ext_vector_type(16)));
typedef _Float16 h8_t __attribute__((ext_vector_type(8)));
typedef _Float16 h4_t __attribute__((ext_vector_type(4)));
typedef float    v8f  __attribute__((ext_vector_type(8)));

// LDS layout (dynamic shared): pos_l f16 [512][128] + preds2_l f16 [512][128] + W2 f16 [128][128] + reduce
#define SM_POS  0
#define SM_P2   (NN * CC * 2)
#define SM_W2   (SM_P2 + NN * CC * 2)
#define SM_RED  (SM_W2 + CC * CC * 2)
#define SM_BYTES (SM_RED + 256 * 4)   // 295,936 B <= 320 KB per workgroup on CDNA5

__device__ __forceinline__ v16h ldfrag(const _Float16* p0, const _Float16* p1) {
  union U { v16h v; h8_t h[2]; } u;
  u.h[0] = *(const h8_t*)p0;
  u.h[1] = *(const h8_t*)p1;
  return u.v;
}

// One fused "logits row-block -> online logsumexp - diag" pass.
// Abase: row-major f16 [512][128] (global for loss1, LDS for loss2). posH: LDS f16 [512][128].
__device__ __forceinline__ float loss_rows(const _Float16* __restrict__ Abase,
                                           const _Float16* posH, int wave, int lane) {
  const int g = lane >> 4, q = lane & 15;
  float part = 0.f;
  for (int rt = 0; rt < 4; ++rt) {
    const int R = (wave * 4 + rt) * 16;
    // A fragments for this 16-row tile, all 4 K-chunks, kept in VGPRs across the m-sweep.
    // 16-bit A 16x32 layout: lane-half g holds K = {g*8..g*8+7} (halves 0-7) and {16+g*8..} (halves 8-15).
    const _Float16* arow = Abase + (size_t)(R + q) * CC;
    v16h a[4];
#pragma unroll
    for (int kk = 0; kk < 4; ++kk) {
      const _Float16* p = arow + kk * 32 + g * 8;
      a[kk] = ldfrag(p, p + 16);
    }
    float mx[8], sm[8];
#pragma unroll
    for (int j = 0; j < 8; ++j) { mx[j] = -1e30f; sm[j] = 0.f; }
    float dval = 0.f;

    for (int mt = 0; mt < 32; ++mt) {
      v8f acc = {};
      // B 32x16 f16: lane-half g holds K = {g*8..g*8+7} and {16+g*8..16+g*8+7} for column N=q.
      const _Float16* brow = posH + (size_t)(mt * 16 + q) * CC + g * 8;
#pragma unroll
      for (int kk = 0; kk < 4; ++kk) {
        const _Float16* p = brow + kk * 32;
        v16h b = ldfrag(p, p + 16);
        acc = __builtin_amdgcn_wmma_f32_16x16x32_f16(false, a[kk], false, b,
                                                     (short)0, acc, false, false);
      }
      if (mt == (R >> 4)) {  // diagonal tile: element (M=r, N=r)
        const int want = (lane < 8) ? lane : (lane - 24);
        float dsel = acc[0];
#pragma unroll
        for (int j = 1; j < 8; ++j) dsel = (want == j) ? acc[j] : dsel;
        dval = ((lane < 8) || (lane >= 24)) ? dsel : 0.f;
      }
      // Online softmax update: exactly one v_exp_f32 per element.
#pragma unroll
      for (int j = 0; j < 8; ++j) {
        const float v  = acc[j];
        const float m0 = mx[j];
        const float M  = fmaxf(m0, v);
        const float e  = __expf(fminf(m0, v) - M);
        sm[j] = (v > m0) ? __fmaf_rn(sm[j], e, 1.0f) : (sm[j] + e);
        mx[j] = M;
      }
    }
    // Butterfly merge of (max,sum) across the 16 lanes of each half (xor<16 never crosses halves).
#pragma unroll
    for (int st = 1; st < 16; st <<= 1) {
#pragma unroll
      for (int j = 0; j < 8; ++j) {
        const float om = __shfl_xor(mx[j], st, 32);
        const float os = __shfl_xor(sm[j], st, 32);
        const float M  = fmaxf(mx[j], om);
        sm[j] = sm[j] * __expf(mx[j] - M) + os * __expf(om - M);
        mx[j] = M;
      }
    }
    float lsum = 0.f;
#pragma unroll
    for (int j = 0; j < 8; ++j) lsum += __logf(sm[j]) + mx[j];
    if (q == 0) part += lsum;  // lane 0 covers rows 0-7, lane 16 covers rows 8-15
    part -= dval;
  }
  return part;
}

extern "C" __global__ void __launch_bounds__(256, 1)
stdim_main(const float* __restrict__ fmap_t, const float* __restrict__ fmap_tp1,
           const float* __restrict__ b2, const _Float16* __restrict__ preds1H,
           const _Float16* __restrict__ w2Hg, float* __restrict__ acc_out) {
  extern __shared__ char smem[];
  _Float16* posH = (_Float16*)(smem + SM_POS);
  _Float16* p2H  = (_Float16*)(smem + SM_P2);
  _Float16* w2H  = (_Float16*)(smem + SM_W2);
  float*    red  = (float*)(smem + SM_RED);

  const int l    = blockIdx.x;
  const int tid  = threadIdx.x;
  const int wave = tid >> 5, lane = tid & 31;
  const int g    = lane >> 4, q = lane & 15;

  // Phase 0: stage pos_l = fmap_tp1[:, l, :] into LDS as f16; copy W2(f16) into LDS.
  for (int c4 = tid; c4 < (NN * CC) / 4; c4 += 256) {
    const int m = c4 >> 5;           // 32 float4 per 128-wide row
    const int off = (c4 & 31) * 4;
    const float4 x = *(const float4*)(fmap_tp1 + ((size_t)m * LN + l) * CC + off);
    h4_t h; h[0] = (_Float16)x.x; h[1] = (_Float16)x.y; h[2] = (_Float16)x.z; h[3] = (_Float16)x.w;
    *(h4_t*)(posH + m * CC + off) = h;
  }
  for (int c8 = tid; c8 < (CC * CC) / 8; c8 += 256)
    *(h8_t*)(w2H + c8 * 8) = *(const h8_t*)(w2Hg + c8 * 8);
  __syncthreads();

  // Phase 1: preds2_l = loc_l @ W2^T + b2 via WMMA (A from global f32 with on-the-fly cvt).
  for (int rt = 0; rt < 4; ++rt) {
    const int R = (wave * 4 + rt) * 16;
    const float* lrow = fmap_t + ((size_t)(R + q) * LN + l) * CC;
    v16h a[4];
#pragma unroll
    for (int kk = 0; kk < 4; ++kk) {
      const int base = kk * 32 + g * 8;
      const float4 x0 = *(const float4*)(lrow + base);
      const float4 x1 = *(const float4*)(lrow + base + 4);
      const float4 y0 = *(const float4*)(lrow + base + 16);
      const float4 y1 = *(const float4*)(lrow + base + 20);
      v16h t;
      t[0]  = (_Float16)x0.x; t[1]  = (_Float16)x0.y; t[2]  = (_Float16)x0.z; t[3]  = (_Float16)x0.w;
      t[4]  = (_Float16)x1.x; t[5]  = (_Float16)x1.y; t[6]  = (_Float16)x1.z; t[7]  = (_Float16)x1.w;
      t[8]  = (_Float16)y0.x; t[9]  = (_Float16)y0.y; t[10] = (_Float16)y0.z; t[11] = (_Float16)y0.w;
      t[12] = (_Float16)y1.x; t[13] = (_Float16)y1.y; t[14] = (_Float16)y1.z; t[15] = (_Float16)y1.w;
      a[kk] = t;
    }
#pragma unroll
    for (int dt = 0; dt < 8; ++dt) {
      v8f acc = {};
#pragma unroll
      for (int kk = 0; kk < 4; ++kk) {
        const _Float16* bp = w2H + (size_t)(dt * 16 + q) * CC + kk * 32 + g * 8;
        v16h b = ldfrag(bp, bp + 16);
        acc = __builtin_amdgcn_wmma_f32_16x16x32_f16(false, a[kk], false, b,
                                                     (short)0, acc, false, false);
      }
      const int col = dt * 16 + q;
      const float bb = b2[col];
#pragma unroll
      for (int j = 0; j < 8; ++j) {
        const int row = R + j + g * 8;   // D layout: VGPR j -> M=j (lanes 0-15) / M=j+8 (lanes 16-31)
        p2H[row * CC + col] = (_Float16)(acc[j] + bb);
      }
    }
  }
  __syncthreads();

  // Phase 2: both fused logits+logsumexp-diag passes.
  float part = loss_rows(preds1H, posH, wave, lane)   // loss1: global-local
             + loss_rows(p2H,     posH, wave, lane);  // loss2: local-local

  red[tid] = part;
  __syncthreads();
  for (int s2 = 128; s2 > 0; s2 >>= 1) {
    if (tid < s2) red[tid] += red[tid + s2];
    __syncthreads();
  }
  if (tid == 0) atomicAdd(acc_out, red[0]);
}

// Prologue: preds1 = f_t @ W1^T + b1 (33 MFLOP, negligible -> plain VALU), W2 -> f16.
extern "C" __global__ void stdim_prep(const float* __restrict__ f_t, const float* __restrict__ W1,
                                      const float* __restrict__ b1, const float* __restrict__ W2,
                                      _Float16* __restrict__ preds1H, _Float16* __restrict__ w2H) {
  const int i = blockIdx.x * blockDim.x + threadIdx.x;
  if (i < NN * CC) {
    const int n = i >> 7, d = i & 127;
    const float* fr = f_t + (size_t)n * DD;
    const float* wr = W1  + (size_t)d * DD;
    float s = 0.f;
#pragma unroll 4
    for (int k = 0; k < DD; k += 4) {
      const float4 aa = *(const float4*)(fr + k);
      const float4 bb = *(const float4*)(wr + k);
      s = __fmaf_rn(aa.x, bb.x, s);
      s = __fmaf_rn(aa.y, bb.y, s);
      s = __fmaf_rn(aa.z, bb.z, s);
      s = __fmaf_rn(aa.w, bb.w, s);
    }
    preds1H[i] = (_Float16)(s + b1[d]);
  }
  if (i < CC * CC) w2H[i] = (_Float16)W2[i];
}

extern "C" __global__ void stdim_zero(float* a) {
  if (blockIdx.x == 0 && threadIdx.x == 0) a[0] = 0.f;
}

extern "C" __global__ void stdim_fin(const float* a, float* out) {
  out[0] = a[0] * (1.0f / ((float)LN * (float)NN));
}

extern "C" void kernel_launch(void* const* d_in, const int* in_sizes, int n_in,
                              void* d_out, int out_size, void* d_ws, size_t ws_size,
                              hipStream_t stream) {
  const float* f_t      = (const float*)d_in[0];
  const float* fmap_t   = (const float*)d_in[1];
  const float* fmap_tp1 = (const float*)d_in[2];
  const float* W1       = (const float*)d_in[3];
  const float* b1       = (const float*)d_in[4];
  const float* W2       = (const float*)d_in[5];
  const float* b2       = (const float*)d_in[6];
  (void)in_sizes; (void)n_in; (void)out_size; (void)ws_size;

  char* ws = (char*)d_ws;
  float*    acc     = (float*)ws;                               // 1 f32 accumulator
  _Float16* preds1H = (_Float16*)(ws + 256);                    // [512][128] f16
  _Float16* w2H     = (_Float16*)(ws + 256 + NN * CC * 2);      // [128][128] f16

  stdim_zero<<<1, 32, 0, stream>>>(acc);
  stdim_prep<<<(NN * CC) / 256, 256, 0, stream>>>(f_t, W1, b1, W2, preds1H, w2H);
  stdim_main<<<LN, 256, SM_BYTES, stream>>>(fmap_t, fmap_tp1, b2, preds1H, w2H, acc);
  stdim_fin<<<1, 1, 0, stream>>>(acc, (float*)d_out);
}
